// MultiHeadSelfAttention_8237747274224
// MI455X (gfx1250) — matrix operand order repaired and checked by reading, not compiled, not run on hardware
//
#include <hip/hip_runtime.h>

// ---------------------------------------------------------------------------
// MHSA for MI455X (gfx1250, wave32). All matmuls on v_wmma_f32_16x16x32_f16.
// B=8, N=1024, D=512, H=8, DH=64.
// ---------------------------------------------------------------------------

typedef _Float16 half_t;
typedef __attribute__((ext_vector_type(16))) _Float16 v16h;
typedef __attribute__((ext_vector_type(8)))  _Float16 v8h;
typedef __attribute__((ext_vector_type(8)))  float    v8f;

constexpr int B_ = 8, N_ = 1024, D_ = 512, H_ = 8, DH_ = 64, BN_ = B_ * N_;

#define WMMA(a, b, c) __builtin_amdgcn_wmma_f32_16x16x32_f16( \
    false, (a), false, (b), (short)0, (c), false, false)

__device__ __forceinline__ v16h cat16(v8h lo, v8h hi) {
  v16h r;
#pragma unroll
  for (int i = 0; i < 8; ++i) { r[i] = lo[i]; r[i + 8] = hi[i]; }
  return r;
}

// A-matrix 16x32 f16 fragment: lane m = lane&15; per lane 16 contiguous
// halfs at k = kbase + 16*grp + e (matches B fragment order).
__device__ __forceinline__ v16h load_a_frag(const half_t* row, int kbase, int grp) {
  v8h lo = *(const v8h*)(row + kbase + 16 * grp);
  v8h hi = *(const v8h*)(row + kbase + 16 * grp + 8);
  return cat16(lo, hi);
}

// B-matrix 32x16 f16 fragment: per lane 16 contiguous halfs (col = lane&15,
// k = 16*grp + e). Caller passes the per-lane base pointer.
__device__ __forceinline__ v16h load_b_frag(const half_t* p) {
  v8h lo = *(const v8h*)p;
  v8h hi = *(const v8h*)(p + 8);
  return cat16(lo, hi);
}

// ---------------------------------------------------------------------------
// Kernel 1: x f32 -> f16 (flat, 8 elems/thread)
// ---------------------------------------------------------------------------
__global__ void cvt_x_kernel(const float* __restrict__ x, half_t* __restrict__ xh) {
  int i = blockIdx.x * blockDim.x + threadIdx.x;      // 0 .. BN_*D_/8 - 1
  const float4* p = (const float4*)x + i * 2;
  float4 a = p[0], b = p[1];
  v8h o;
  o[0] = (half_t)a.x; o[1] = (half_t)a.y; o[2] = (half_t)a.z; o[3] = (half_t)a.w;
  o[4] = (half_t)b.x; o[5] = (half_t)b.y; o[6] = (half_t)b.z; o[7] = (half_t)b.w;
  *(v8h*)(xh + i * 8) = o;
}

// ---------------------------------------------------------------------------
// Kernel 2: W [in,out] f32 -> Wt [out,in] f16  (blockIdx.y selects Q/K/V)
// ---------------------------------------------------------------------------
__global__ void transpose_w_kernel(const float* __restrict__ Wq,
                                   const float* __restrict__ Wk,
                                   const float* __restrict__ Wv,
                                   half_t* __restrict__ Wt) {
  int mat = blockIdx.y;
  const float* W = (mat == 0) ? Wq : (mat == 1) ? Wk : Wv;
  half_t* dst = Wt + (size_t)mat * D_ * D_;
  int idx = blockIdx.x * blockDim.x + threadIdx.x;    // 0 .. D_*D_-1
  int o = idx & (D_ - 1);
  int i = idx >> 9;
  dst[o * D_ + i] = (half_t)W[i * D_ + o];            // coalesced read
}

// ---------------------------------------------------------------------------
// Kernel 3: QKV projection GEMM.  out = x @ W + b   ([BN,512] x [512,512])
// Block: 256 thr = 8 waves; wave tile 16 rows x 64 cols; block tile 128x64.
// Grid: (BN/128=64, D/64=8, 3). Q,K stored [b,h,tok,d] f16; V stored
// transposed [b,h,d,tok] f16 (one packed b128 store per C fragment).
// ---------------------------------------------------------------------------
__global__ __launch_bounds__(256) void qkv_proj_kernel(
    const half_t* __restrict__ xh, const half_t* __restrict__ Wt,
    const float* __restrict__ bq, const float* __restrict__ bk,
    const float* __restrict__ bv,
    half_t* __restrict__ Q, half_t* __restrict__ K, half_t* __restrict__ Vt) {
  const int mat  = blockIdx.z;
  const half_t* W    = Wt + (size_t)mat * D_ * D_;
  const float*  bias = (mat == 0) ? bq : (mat == 1) ? bk : bv;

  const int wave = threadIdx.x >> 5;
  const int lane = threadIdx.x & 31;
  const int grp  = lane >> 4;
  const int lan  = lane & 15;

  const int rowbase = blockIdx.x * 128 + wave * 16;   // token rows (flat BN)
  const int colbase = blockIdx.y * 64;                // output features

  v8f acc[4] = {};

  const half_t* arow = xh + (size_t)(rowbase + lan) * D_;
  const half_t* brow0 = W + (size_t)(colbase +  0 + lan) * D_ + 16 * grp;
  const half_t* brow1 = W + (size_t)(colbase + 16 + lan) * D_ + 16 * grp;
  const half_t* brow2 = W + (size_t)(colbase + 32 + lan) * D_ + 16 * grp;
  const half_t* brow3 = W + (size_t)(colbase + 48 + lan) * D_ + 16 * grp;

  for (int k = 0; k < D_; k += 32) {
    v16h a  = load_a_frag(arow, k, grp);
    v16h b0 = load_b_frag(brow0 + k);
    v16h b1 = load_b_frag(brow1 + k);
    v16h b2 = load_b_frag(brow2 + k);
    v16h b3 = load_b_frag(brow3 + k);
    acc[0] = WMMA(a, b0, acc[0]);
    acc[1] = WMMA(a, b1, acc[1]);
    acc[2] = WMMA(a, b2, acc[2]);
    acc[3] = WMMA(a, b3, acc[3]);
  }

  const int b = rowbase >> 10;          // batch (128-row block never straddles)
  const int h = blockIdx.y;             // colbase == h*64 exactly
  const int ntokbase = rowbase & (N_ - 1);

  if (mat < 2) {
    half_t* dst = (mat == 0) ? Q : K;   // layout [b,h,tok,d]
#pragma unroll
    for (int j = 0; j < 4; ++j) {
      const int d = 16 * j + lan;
      const float bj = bias[h * DH_ + d];
#pragma unroll
      for (int r = 0; r < 8; ++r) {
        const int tok = ntokbase + r + 8 * grp;
        dst[(((size_t)(b * H_ + h) * N_) + tok) * DH_ + d] =
            (half_t)(acc[j][r] + bj);
      }
    }
  } else {                              // V transposed: [b,h,d,tok]
#pragma unroll
    for (int j = 0; j < 4; ++j) {
      const int d = 16 * j + lan;
      const float bj = bias[h * DH_ + d];
      v8h pk;
#pragma unroll
      for (int r = 0; r < 8; ++r) pk[r] = (half_t)(acc[j][r] + bj);
      const int tok0 = ntokbase + 8 * grp;   // 8 consecutive tokens
      *(v8h*)(Vt + ((size_t)(b * H_ + h) * DH_ + d) * N_ + tok0) = pk;
    }
  }
}

// ---------------------------------------------------------------------------
// Kernel 4: flash attention per (b,h). Block = 8 waves x 16 query rows = 128
// rows. Grid (B*H=64, N/128=8). Key tiles of 32. Online softmax (no 1/sqrt
// scaling, matching the reference). P: C-layout -> A-layout via per-wave LDS
// bounce with explicit s_wait_dscnt.
// ---------------------------------------------------------------------------
__global__ __launch_bounds__(256) void attn_kernel(
    const half_t* __restrict__ Q, const half_t* __restrict__ K,
    const half_t* __restrict__ Vt, float* __restrict__ out) {
  __shared__ __align__(32) half_t lds[8 * 16 * 32];   // 1KB per wave

  const int bh   = blockIdx.x;
  const int qblk = blockIdx.y;
  const int wave = threadIdx.x >> 5;
  const int lane = threadIdx.x & 31;
  const int grp  = lane >> 4;
  const int lan  = lane & 15;
  const int b = bh >> 3, h = bh & 7;
  const int qbase = qblk * 128 + wave * 16;

  // Q A-fragments held for the whole loop (16 rows x 64 = 2 frags)
  const half_t* qrow = Q + ((size_t)bh * N_ + qbase + lan) * DH_;
  const v16h aq0 = load_a_frag(qrow, 0, grp);
  const v16h aq1 = load_a_frag(qrow, 32, grp);

  v8f o[4] = {};
  float mrun[8], srun[8];
#pragma unroll
  for (int r = 0; r < 8; ++r) { mrun[r] = -3.0e38f; srun[r] = 0.0f; }

  const half_t* kb  = K  + (size_t)bh * N_ * DH_ + (size_t)lan * DH_ + 16 * grp;
  const half_t* vb0 = Vt + ((size_t)bh * DH_ +  0 + lan) * N_ + 16 * grp;
  const half_t* vb1 = Vt + ((size_t)bh * DH_ + 16 + lan) * N_ + 16 * grp;
  const half_t* vb2 = Vt + ((size_t)bh * DH_ + 32 + lan) * N_ + 16 * grp;
  const half_t* vb3 = Vt + ((size_t)bh * DH_ + 48 + lan) * N_ + 16 * grp;
  half_t* wl = lds + wave * 512;

  for (int kt = 0; kt < N_; kt += 32) {
    // ---- S = Q @ K^T for 32 keys (two 16-col C tiles) ----
    v16h bk00 = load_b_frag(kb + (size_t)kt * DH_);
    v16h bk01 = load_b_frag(kb + (size_t)kt * DH_ + 32);
    v16h bk10 = load_b_frag(kb + (size_t)(kt + 16) * DH_);
    v16h bk11 = load_b_frag(kb + (size_t)(kt + 16) * DH_ + 32);
    v8f s0 = {}, s1 = {};
    s0 = WMMA(aq0, bk00, s0);
    s0 = WMMA(aq1, bk01, s0);
    s1 = WMMA(aq0, bk10, s1);
    s1 = WMMA(aq1, bk11, s1);

    // ---- online softmax (row = 16 lanes of one half-wave, VGPR r) ----
#pragma unroll
    for (int r = 0; r < 8; ++r) {
      float t = fmaxf(s0[r], s1[r]);
      t = fmaxf(t, __shfl_xor(t, 1, 32));
      t = fmaxf(t, __shfl_xor(t, 2, 32));
      t = fmaxf(t, __shfl_xor(t, 4, 32));
      t = fmaxf(t, __shfl_xor(t, 8, 32));
      const float nm    = fmaxf(mrun[r], t);
      const float alpha = __expf(mrun[r] - nm);
      mrun[r] = nm;
      const float p0 = __expf(s0[r] - nm);
      const float p1 = __expf(s1[r] - nm);
      s0[r] = p0; s1[r] = p1;
      float rs = p0 + p1;
      rs += __shfl_xor(rs, 1, 32);
      rs += __shfl_xor(rs, 2, 32);
      rs += __shfl_xor(rs, 4, 32);
      rs += __shfl_xor(rs, 8, 32);
      srun[r] = srun[r] * alpha + rs;
      o[0][r] *= alpha; o[1][r] *= alpha; o[2][r] *= alpha; o[3][r] *= alpha;
    }

    // ---- P (f32 C layout) -> f16 A layout, via per-wave LDS bounce ----
#pragma unroll
    for (int r = 0; r < 8; ++r) {
      const int m = r + 8 * grp;
      wl[m * 32 + lan]      = (half_t)s0[r];
      wl[m * 32 + 16 + lan] = (half_t)s1[r];
    }
    asm volatile("s_wait_dscnt 0" ::: "memory");   // same-wave DS store->load
    v8h plo = *(const v8h*)(wl + lan * 32 + 16 * grp);
    v8h phi = *(const v8h*)(wl + lan * 32 + 16 * grp + 8);
    const v16h pA = cat16(plo, phi);
    asm volatile("" ::: "memory");

    // ---- O += P @ V (V stored transposed: contiguous B fragments) ----
    v16h bv0 = load_b_frag(vb0 + kt);
    v16h bv1 = load_b_frag(vb1 + kt);
    v16h bv2 = load_b_frag(vb2 + kt);
    v16h bv3 = load_b_frag(vb3 + kt);
    o[0] = WMMA(pA, bv0, o[0]);
    o[1] = WMMA(pA, bv1, o[1]);
    o[2] = WMMA(pA, bv2, o[2]);
    o[3] = WMMA(pA, bv3, o[3]);
  }

  // ---- normalize + write f32 output [b, tok, h*64+d] ----
  float inv[8];
#pragma unroll
  for (int r = 0; r < 8; ++r) inv[r] = 1.0f / srun[r];
#pragma unroll
  for (int j = 0; j < 4; ++j) {
    const int d = 16 * j + lan;
#pragma unroll
    for (int r = 0; r < 8; ++r) {
      const int q = qbase + r + 8 * grp;
      out[((size_t)(b * N_ + q)) * D_ + h * DH_ + d] = o[j][r] * inv[r];
    }
  }
}

// ---------------------------------------------------------------------------
// Launch
// ---------------------------------------------------------------------------
extern "C" void kernel_launch(void* const* d_in, const int* in_sizes, int n_in,
                              void* d_out, int out_size, void* d_ws, size_t ws_size,
                              hipStream_t stream) {
  const float* x  = (const float*)d_in[0];
  const float* Wq = (const float*)d_in[1];
  const float* bq = (const float*)d_in[2];
  const float* Wk = (const float*)d_in[3];
  const float* bk = (const float*)d_in[4];
  const float* Wv = (const float*)d_in[5];
  const float* bv = (const float*)d_in[6];
  float* out = (float*)d_out;

  // Workspace layout (f16 intermediates), ~33.5 MB total
  char* ws = (char*)d_ws;
  const size_t XH_BYTES = (size_t)BN_ * D_ * 2;        // 8 MB
  const size_t WT_BYTES = (size_t)3 * D_ * D_ * 2;     // 1.5 MB
  const size_t QK_BYTES = (size_t)BN_ * D_ * 2;        // 8 MB each
  half_t* xh = (half_t*)(ws);
  half_t* Wt = (half_t*)(ws + XH_BYTES);
  half_t* Q  = (half_t*)(ws + XH_BYTES + WT_BYTES);
  half_t* K  = (half_t*)(ws + XH_BYTES + WT_BYTES + QK_BYTES);
  half_t* Vt = (half_t*)(ws + XH_BYTES + WT_BYTES + 2 * QK_BYTES);

  cvt_x_kernel<<<(BN_ * D_ / 8) / 256, 256, 0, stream>>>(x, xh);
  transpose_w_kernel<<<dim3(D_ * D_ / 256, 3), 256, 0, stream>>>(Wq, Wk, Wv, Wt);
  qkv_proj_kernel<<<dim3(BN_ / 128, D_ / 64, 3), 256, 0, stream>>>(
      xh, Wt, bq, bk, bv, Q, K, Vt);
  attn_kernel<<<dim3(B_ * H_, N_ / 128), 256, 0, stream>>>(Q, K, Vt, out);
}
